// MultiHeadAttention_20349555048623
// MI455X (gfx1250) — matrix operand order repaired and checked by reading, not compiled, not run on hardware
//
#include <hip/hip_runtime.h>
#include <hip/hip_bf16.h>

typedef __bf16 bf16_t;
typedef __attribute__((ext_vector_type(16))) __bf16 v16bf;
typedef __attribute__((ext_vector_type(8)))  __bf16 v8bf;
typedef __attribute__((ext_vector_type(8)))  float  v8f;

#define N_EMBD 768
#define SEQ    4096
#define NBATCH 2
#define NH     12
#define HD     64
#define NTOK   (NBATCH*SEQ)     // 8192 rows
#define KT16   (SEQ/16)         // 256 key tiles of 16
#define KB32   (SEQ/32)         // 128 key blocks of 32
#define CB     (N_EMBD/32)      // 24 K-chunks in projections
#define JT     (N_EMBD/16)      // 48 column tiles

static __device__ inline v8f wmma_bf16(v16bf a, v16bf b, v8f c) {
  // D = A(16x32 bf16) x B(32x16 bf16) + C(16x16 f32)
  return __builtin_amdgcn_wmma_f32_16x16x32_bf16(false, a, false, b, (short)0, c, false, false);
}

// A-fragment (16x32 bf16) from row-major memory; lane holds row M=lane&15,
// K = kb..kb+15 with kb = (lane>>4)*16  (contiguous, matches packed B)
static __device__ inline v16bf load_a_frag(const bf16_t* base, int ld) {
  int lane = threadIdx.x & 31;
  int m  = lane & 15;
  int kb = (lane >> 4) * 16;
  const bf16_t* p = base + m * ld + kb;
  v8bf lo = *(const v8bf*)(p);
  v8bf hi = *(const v8bf*)(p + 8);
  return __builtin_shufflevector(lo, hi, 0,1,2,3,4,5,6,7,8,9,10,11,12,13,14,15);
}

// CDNA5 async copy: 16 bytes global -> LDS, tracked by ASYNCcnt
static __device__ inline void async_copy16(const bf16_t* lds_dst, const bf16_t* gsrc) {
  unsigned loff = (unsigned)(unsigned long long)lds_dst;      // low 32 bits = LDS addr
  unsigned long long ga = (unsigned long long)gsrc;
  asm volatile("global_load_async_to_lds_b128 %0, %1, off"
               :: "v"(loff), "v"(ga) : "memory");
}

// ---------------- pack kernels ----------------
__global__ void k_pack_x(const float* __restrict__ x, bf16_t* __restrict__ xb, int n) {
  int i = blockIdx.x * 256 + threadIdx.x;
  if (i < n) xb[i] = (bf16_t)x[i];
}

// Pack W[j][c] (torch Linear, y = x @ W^T) into B-fragment order:
// out[((jb*CB+cb)*32 + lane)*16 + i] = W[jb*16 + (lane&15)][cb*32 + (lane>>4)*16 + i]
__global__ void k_pack_w(const float* __restrict__ w0, const float* __restrict__ w1,
                         const float* __restrict__ w2, const float* __restrict__ w3,
                         bf16_t* __restrict__ o0, bf16_t* __restrict__ o1,
                         bf16_t* __restrict__ o2, bf16_t* __restrict__ o3) {
  const float* w; bf16_t* o;
  switch (blockIdx.y) {
    case 0:  w = w0; o = o0; break;
    case 1:  w = w1; o = o1; break;
    case 2:  w = w2; o = o2; break;
    default: w = w3; o = o3; break;
  }
  int p = blockIdx.x * 256 + threadIdx.x;      // 0 .. 768*768-1
  int i    = p & 15;
  int lane = (p >> 4) & 31;
  int cb   = (p >> 9) % CB;
  int jb   = p / (512 * CB);
  int j = jb * 16 + (lane & 15);
  int c = cb * 32 + (lane >> 4) * 16 + i;
  o[p] = (bf16_t)w[j * N_EMBD + c];
}

// ---------------- fused QKV projection ----------------
// grid(NTOK/16, 36), block 128 (4 waves). wave -> (which in {q,k,v}, jt)
__global__ void __launch_bounds__(128) k_qkv(
    const bf16_t* __restrict__ xb,
    const bf16_t* __restrict__ wqp, const bf16_t* __restrict__ wkp, const bf16_t* __restrict__ wvp,
    const float* __restrict__ bq, const float* __restrict__ bk, const float* __restrict__ bv,
    bf16_t* __restrict__ Qb, bf16_t* __restrict__ Kp, bf16_t* __restrict__ Vp) {
  int lane = threadIdx.x & 31;
  int wave = threadIdx.x >> 5;
  int rt   = blockIdx.x;
  int flat = blockIdx.y * 4 + wave;     // 0..143
  int which = flat / JT;                // 0=Q 1=K 2=V
  int jt    = flat % JT;
  const bf16_t* w    = (which == 0) ? wqp : (which == 1) ? wkp : wvp;
  const float*  bias = (which == 0) ? bq  : (which == 1) ? bk  : bv;

  v8f acc = {};
  const bf16_t* abase = xb + (size_t)rt * 16 * N_EMBD;
  #pragma unroll 4
  for (int cb = 0; cb < CB; ++cb) {
    v16bf a = load_a_frag(abase + cb * 32, N_EMBD);
    v16bf b = *(const v16bf*)(w + ((size_t)(jt * CB + cb) * 32 + lane) * 16);
    acc = wmma_bf16(a, b, acc);
  }

  int col = lane & 15, hi = lane >> 4;
  int j = jt * 16 + col, h = j >> 6, d = j & 63;
  float bval = bias[j];
  #pragma unroll
  for (int r = 0; r < 8; ++r) {
    int grow = rt * 16 + r + 8 * hi;
    int bb = grow >> 12, tt = grow & (SEQ - 1);
    int bh = bb * NH + h;
    float v = acc[r] + bval;
    if (which == 0) {
      Qb[((size_t)bh * SEQ + tt) * HD + d] = (bf16_t)v;              // [B,H,T,D]
    } else if (which == 1) {
      // K^T B-fragment pack: tile (kt, db): lane = lh*16 + (key&15), elem = d&15
      int kt = tt >> 4, nn = tt & 15, db = d >> 5, lh = (d >> 4) & 1, ii = d & 15;
      Kp[((size_t)((bh * KT16 + kt) * 2 + db) * 32 + lh * 16 + nn) * 16 + ii] = (bf16_t)v;
    } else {
      // V B-fragment pack: tile (kb, dt): lane = lh*16 + (d&15), elem = key&15
      int kb = tt >> 5, lh = (tt >> 4) & 1, ii = tt & 15, dt = d >> 4, nn = d & 15;
      Vp[((size_t)((bh * KB32 + kb) * 4 + dt) * 32 + lh * 16 + nn) * 16 + ii] = (bf16_t)v;
    }
  }
}

// ---------------- flash attention ----------------
// block = 128 threads = 4 waves, one (b,h), 64 consecutive queries.
// K/V chunks (32 keys, 8KB packed fragments) staged once per block into LDS
// via async global->LDS copies, double-buffered.
__global__ void __launch_bounds__(128) k_attn(
    const bf16_t* __restrict__ Qb, const bf16_t* __restrict__ Kp,
    const bf16_t* __restrict__ Vp, bf16_t* __restrict__ Yb) {
  __shared__ bf16_t kvbuf[2][4096];   // [buf][ K: 4*512 | V: 4*512 ]  = 16KB
  __shared__ bf16_t pbuf[4][512];     // per-wave P-tile staging        = 4KB
  int lane = threadIdx.x & 31;
  int wave = threadIdx.x >> 5;
  int col = lane & 15, hi = lane >> 4;
  int h = blockIdx.y, bz = blockIdx.z;
  int bh = bz * NH + h;
  int q0 = blockIdx.x * 64 + wave * 16;
  int nc_w   = (q0 + 47) >> 5;        // chunks this wave needs (causal)
  int nc_blk = 2 * blockIdx.x + 2;    // chunks the block stages

  const bf16_t* Kbase = Kp + (size_t)bh * (KT16 * 2 * 512);
  const bf16_t* Vbase = Vp + (size_t)bh * (KB32 * 4 * 512);

  const bf16_t* qbase = Qb + ((size_t)bh * SEQ + q0) * HD;
  v16bf aq0 = load_a_frag(qbase, HD);        // d = 0..31
  v16bf aq1 = load_a_frag(qbase + 32, HD);   // d = 32..63

  v8f acc0 = {}, acc1 = {}, acc2 = {}, acc3 = {};
  float m_r[8], l_r[8];
  #pragma unroll
  for (int r = 0; r < 8; ++r) { m_r[r] = -1e30f; l_r[r] = 0.0f; }

  // stage chunk c: 4KB K-fragments + 4KB V-fragments -> kvbuf[c&1]
  auto stage = [&](int c) {
    const bf16_t* ks = Kbase + (size_t)c * 2048;
    const bf16_t* vs = Vbase + (size_t)c * 2048;
    bf16_t* dst = kvbuf[c & 1];
    int t8 = threadIdx.x * 8;                       // 16B per thread per piece
    async_copy16(dst + t8,        ks + t8);
    async_copy16(dst + 1024 + t8, ks + 1024 + t8);
    async_copy16(dst + 2048 + t8, vs + t8);
    async_copy16(dst + 3072 + t8, vs + 1024 + t8);
  };

  stage(0);
  for (int c = 0; c < nc_blk; ++c) {
    asm volatile("s_wait_asynccnt 0" ::: "memory");   // my stage(c) landed
    __syncthreads();                                  // everyone's landed; c-1 consumed
    if (c + 1 < nc_blk) stage(c + 1);                 // prefetch into other buffer

    if (c < nc_w) {
      const bf16_t* kc = kvbuf[c & 1];
      v16bf bk0 = *(const v16bf*)(kc + lane * 16);          // ktA, d 0..31
      v16bf bk1 = *(const v16bf*)(kc + 512  + lane * 16);   // ktA, d 32..63
      v16bf bk2 = *(const v16bf*)(kc + 1024 + lane * 16);   // ktB, d 0..31
      v16bf bk3 = *(const v16bf*)(kc + 1536 + lane * 16);   // ktB, d 32..63
      v8f s0 = {}, s1 = {};
      s0 = wmma_bf16(aq0, bk0, s0);
      s0 = wmma_bf16(aq1, bk1, s0);
      s1 = wmma_bf16(aq0, bk2, s1);
      s1 = wmma_bf16(aq1, bk3, s1);

      bool domask = (c * 32 + 31 > q0);
      float p0[8], p1[8];
      #pragma unroll
      for (int r = 0; r < 8; ++r) {
        float v0 = s0[r] * 0.125f, v1 = s1[r] * 0.125f;   // 1/sqrt(64)
        if (domask) {
          int row = q0 + r + 8 * hi;
          if (c * 32 + col > row)       v0 = -1e30f;
          if (c * 32 + 16 + col > row)  v1 = -1e30f;
        }
        float mx = fmaxf(v0, v1);
        #pragma unroll
        for (int sh = 8; sh >= 1; sh >>= 1) mx = fmaxf(mx, __shfl_xor(mx, sh, 16));
        float mnew  = fmaxf(m_r[r], mx);
        float alpha = __expf(m_r[r] - mnew);
        float e0 = __expf(v0 - mnew), e1 = __expf(v1 - mnew);
        float sum = e0 + e1;
        #pragma unroll
        for (int sh = 8; sh >= 1; sh >>= 1) sum += __shfl_xor(sum, sh, 16);
        l_r[r] = l_r[r] * alpha + sum;
        m_r[r] = mnew;
        acc0[r] *= alpha; acc1[r] *= alpha; acc2[r] *= alpha; acc3[r] *= alpha;
        p0[r] = e0; p1[r] = e1;
      }

      // C-layout -> row-major LDS (bf16) -> A-layout fragment (wave-private)
      bf16_t* pb = pbuf[wave];
      #pragma unroll
      for (int r = 0; r < 8; ++r) {
        int M = r + 8 * hi;
        pb[M * 32 + col]      = (bf16_t)p0[r];
        pb[M * 32 + 16 + col] = (bf16_t)p1[r];
      }
      asm volatile("s_wait_dscnt 0" ::: "memory");   // wave-local LDS RAW fence
      int m2 = lane & 15, kb = hi * 16;
      v8bf plo = *(const v8bf*)&pb[m2 * 32 + kb];
      v8bf phi = *(const v8bf*)&pb[m2 * 32 + kb + 8];
      v16bf ap = __builtin_shufflevector(plo, phi, 0,1,2,3,4,5,6,7,8,9,10,11,12,13,14,15);

      const bf16_t* vc = kvbuf[c & 1] + 2048;
      v16bf bv0 = *(const v16bf*)(vc + lane * 16);
      v16bf bv1 = *(const v16bf*)(vc + 512  + lane * 16);
      v16bf bv2 = *(const v16bf*)(vc + 1024 + lane * 16);
      v16bf bv3 = *(const v16bf*)(vc + 1536 + lane * 16);
      acc0 = wmma_bf16(ap, bv0, acc0);
      acc1 = wmma_bf16(ap, bv1, acc1);
      acc2 = wmma_bf16(ap, bv2, acc2);
      acc3 = wmma_bf16(ap, bv3, acc3);
    }
  }

  #pragma unroll
  for (int r = 0; r < 8; ++r) {
    float rl = 1.0f / l_r[r];
    int qa = q0 + r + 8 * hi;
    bf16_t* yrow = Yb + ((size_t)bz * SEQ + qa) * N_EMBD + h * HD + col;
    yrow[0]  = (bf16_t)(acc0[r] * rl);
    yrow[16] = (bf16_t)(acc1[r] * rl);
    yrow[32] = (bf16_t)(acc2[r] * rl);
    yrow[48] = (bf16_t)(acc3[r] * rl);
  }
}

// ---------------- output projection (f32 out + bias) ----------------
__global__ void __launch_bounds__(128) k_oproj(
    const bf16_t* __restrict__ Yb, const bf16_t* __restrict__ wop,
    const float* __restrict__ bo, float* __restrict__ out) {
  int lane = threadIdx.x & 31;
  int wave = threadIdx.x >> 5;
  int rt = blockIdx.x;
  int jt = blockIdx.y * 4 + wave;      // 0..47
  v8f acc = {};
  const bf16_t* abase = Yb + (size_t)rt * 16 * N_EMBD;
  #pragma unroll 4
  for (int cb = 0; cb < CB; ++cb) {
    v16bf a = load_a_frag(abase + cb * 32, N_EMBD);
    v16bf b = *(const v16bf*)(wop + ((size_t)(jt * CB + cb) * 32 + lane) * 16);
    acc = wmma_bf16(a, b, acc);
  }
  int col = lane & 15, hi = lane >> 4;
  float bval = bo[jt * 16 + col];
  #pragma unroll
  for (int r = 0; r < 8; ++r) {
    int grow = rt * 16 + r + 8 * hi;
    out[(size_t)grow * N_EMBD + jt * 16 + col] = acc[r] + bval;
  }
}

// ---------------- host ----------------
extern "C" void kernel_launch(void* const* d_in, const int* in_sizes, int n_in,
                              void* d_out, int out_size, void* d_ws, size_t ws_size,
                              hipStream_t stream) {
  const float* x  = (const float*)d_in[0];
  const float* Wq = (const float*)d_in[1];
  const float* bq = (const float*)d_in[2];
  const float* Wk = (const float*)d_in[3];
  const float* bk = (const float*)d_in[4];
  const float* Wv = (const float*)d_in[5];
  const float* bv = (const float*)d_in[6];
  const float* Wo = (const float*)d_in[7];
  const float* bo = (const float*)d_in[8];

  char* ws = (char*)d_ws;
  size_t off = 0;
  auto carve = [&](size_t bytes) {
    void* p = ws + off;
    off = (off + bytes + 255) & ~(size_t)255;
    return p;
  };
  bf16_t* xb  = (bf16_t*)carve((size_t)NTOK * N_EMBD * 2);    // 12.6 MB
  bf16_t* wqp = (bf16_t*)carve((size_t)N_EMBD * N_EMBD * 2);  // 1.2 MB each
  bf16_t* wkp = (bf16_t*)carve((size_t)N_EMBD * N_EMBD * 2);
  bf16_t* wvp = (bf16_t*)carve((size_t)N_EMBD * N_EMBD * 2);
  bf16_t* wop = (bf16_t*)carve((size_t)N_EMBD * N_EMBD * 2);
  bf16_t* Qb  = (bf16_t*)carve((size_t)NTOK * N_EMBD * 2);
  bf16_t* Kp  = (bf16_t*)carve((size_t)NTOK * N_EMBD * 2);
  bf16_t* Vp  = (bf16_t*)carve((size_t)NTOK * N_EMBD * 2);
  bf16_t* Yb  = (bf16_t*)carve((size_t)NTOK * N_EMBD * 2);    // total ~67.6 MB

  k_pack_x<<<(NTOK * N_EMBD) / 256, 256, 0, stream>>>(x, xb, NTOK * N_EMBD);
  k_pack_w<<<dim3((N_EMBD * N_EMBD) / 256, 4), 256, 0, stream>>>(Wq, Wk, Wv, Wo,
                                                                 wqp, wkp, wvp, wop);
  k_qkv<<<dim3(NTOK / 16, 36), 128, 0, stream>>>(xb, wqp, wkp, wvp, bq, bk, bv, Qb, Kp, Vp);
  k_attn<<<dim3(SEQ / 64, NH, NBATCH), 128, 0, stream>>>(Qb, Kp, Vp, Yb);
  k_oproj<<<dim3(NTOK / 16, 12), 128, 0, stream>>>(Yb, wop, bo, (float*)d_out);
}
